// Decoder_42460046688705
// MI455X (gfx1250) — matrix operand order repaired and checked by reading, not compiled, not run on hardware
//
#include <hip/hip_runtime.h>

// ---------------- problem constants ----------------
#define D_   1024
#define T_   1024
#define H_   16
#define HS_  64
#define L_   4
#define B_   4
#define FF_  4096
#define V_   32000
#define NT_  (B_ * T_)      // 4096 tokens
#define CHUNK_ 3200         // vocab chunk (multiple of 64), 10 chunks
#define NCHUNK_ 10

typedef __attribute__((ext_vector_type(16))) _Float16 v16h;
typedef __attribute__((ext_vector_type(8)))  _Float16 v8h;
typedef __attribute__((ext_vector_type(8)))  float    v8f;
typedef unsigned int u32x4 __attribute__((ext_vector_type(4)));
typedef int          i32x4 __attribute__((ext_vector_type(4)));
typedef int          i32x8 __attribute__((ext_vector_type(8)));

#define CONCAT16(lo, hi) \
    __builtin_shufflevector(lo, hi, 0,1,2,3,4,5,6,7,8,9,10,11,12,13,14,15)

// Tensor Data Mover: issue one 2D tile DMA global->LDS.
// clang-22 (ROCm 7.2): 5-arg builtin; clang-23 (amdgpu-toolchain): 6-arg.
__device__ static inline void tdm_load(u32x4 g0, i32x8 g1)
{
#if defined(__clang_major__) && (__clang_major__ >= 23)
    __builtin_amdgcn_tensor_load_to_lds(g0, g1, (i32x4)0, (i32x4)0, (i32x8)0, 0);
#else
    __builtin_amdgcn_tensor_load_to_lds(g0, g1, (i32x4)0, (i32x4)0, 0);
#endif
}

// ============================================================
// Generic WMMA GEMM:  C = act(A(f16)[MxK] * B(f16)[KxN] + bias + resid)
//  block tile 128x64, 8 waves (wave32), each wave 2x2 16x16 tiles.
//  Double-buffered software pipeline:
//    - A tile (128x32) via TDM tensor_load_to_lds into alternate buffer,
//      issued one k-step ahead, drained with s_wait_tensorcnt before use.
//    - B tile staged column-major [64][40]; next tile's global loads are
//      issued before the WMMA block so they fly during the math.
//  flags: 1=transB  2=outF16  4=relu  8=bias  16=resid
//  batch z decomposed as (z/H_, z%H_) with two strides per operand.
// ============================================================
__global__ __launch_bounds__(256) void k_gemm(
    const _Float16* __restrict__ A, const _Float16* __restrict__ B,
    void* __restrict__ Cv,
    const float* __restrict__ bias, const float* __restrict__ resid,
    int lda, int ldb, int ldc, int K,
    long sA1, long sA2, long sB1, long sB2, long sC1, long sC2,
    int flags)
{
    const int transB = flags & 1, outF16 = flags & 2, doRelu = flags & 4,
              hasBias = flags & 8, hasRes = flags & 16;

    __shared__ __align__(16) _Float16 As[2][128 * 40];  // 128x32, pitch 40
    __shared__ __align__(16) _Float16 Bs[2][64 * 40];   // col-major [n][k], pitch 40

    const int z = blockIdx.z;
    const long offA = (long)(z / H_) * sA1 + (long)(z % H_) * sA2;
    const long offB = (long)(z / H_) * sB1 + (long)(z % H_) * sB2;
    const long offC = (long)(z / H_) * sC1 + (long)(z % H_) * sC2;

    const int bm0 = blockIdx.x * 128;
    const int bn0 = blockIdx.y * 64;
    const int tid  = threadIdx.x;
    const int lane = tid & 31;
    const int wid  = tid >> 5;
    const int wm   = wid & 3;   // 4 wave-rows  (32 rows each)
    const int wn   = wid >> 2;  // 2 wave-cols  (32 cols each)
    const int ka   = (lane < 16) ? 0 : 8;    // A-frag k base (ISA 7.12.2)
    const int kb   = (lane < 16) ? 0 : 8;    // B-frag k base

    // ---- per-thread B staging geometry ----
    // transB : Bg[n*ldb + k]  -> vector load along k, vector LDS store
    // !transB: Bg[k*ldb + n]  -> coalesced load along n, LDS scatter
    const int bn_ = transB ? (tid >> 2) : 0;           // n row (transB)
    const int bks = transB ? ((tid & 3) * 8) : 0;      // k seg  (transB)
    const int bk_ = transB ? 0 : (tid >> 3);           // k row  (!transB)
    const int bc_ = transB ? 0 : ((tid & 7) * 8);      // n seg  (!transB)
    const _Float16* bptr = transB
        ? (B + offB + (long)(bn0 + bn_) * ldb + bks)
        : (B + offB + (long)bk_ * ldb + bn0 + bc_);
    const long bstep = transB ? 1 : (long)ldb;         // elem step per +1 k

    // ---- TDM descriptor (wave 0 only; global/lds addr updated per issue) ----
    u32x4 g0 = (u32x4)0;
    i32x8 g1 = (i32x8)0;
    unsigned long long gbase = 0;
    unsigned lds0 = 0, lds1 = 0;
    if (wid == 0) {
        gbase = (unsigned long long)(uintptr_t)(A + offA + (long)bm0 * lda);
        lds0 = (unsigned)(uintptr_t)(&As[0][0]);
        lds1 = (unsigned)(uintptr_t)(&As[1][0]);
        g0[0] = 1u;                                   // count=1, user desc
        // data_size=2B (code 1), pad_enable, pad_interval=16 DW, pad_amount=4 DW
        g1[0] = (int)((1u << 16) | (1u << 20) | (3u << 22) | (3u << 25));
        const unsigned TD = 1u << 30;                 // huge tensor dims: no OOB clip
        g1[1] = (int)((TD & 0xffffu) << 16);                           // td0[15:0]
        g1[2] = (int)(((TD >> 16) & 0xffffu) | ((TD & 0xffffu) << 16));// td0[31:16]|td1[15:0]
        g1[3] = (int)(((TD >> 16) & 0xffffu) | (32u << 16));           // td1[31:16]|tile0=32
        g1[4] = 128;                                  // tile1=128, tile2=0
        g1[5] = (int)(unsigned)lda;                   // dim0 stride (elements)
        g1[6] = 0;
        g1[7] = 0;
    }

    // ---- prologue: stage tile 0 into buffer 0 ----
    {
        v8h breg = *(const v8h*)bptr;
        if (wid == 0) {
            g0[1] = lds0;
            g0[2] = (unsigned)(gbase & 0xffffffffull);
            g0[3] = (unsigned)((gbase >> 32) & 0x1ffffffull) | (2u << 30); // type=2
            tdm_load(g0, g1);
        }
        if (transB) {
            *(v8h*)&Bs[0][bn_ * 40 + bks] = breg;
        } else {
#pragma unroll
            for (int i = 0; i < 8; ++i) Bs[0][(bc_ + i) * 40 + bk_] = breg[i];
        }
        if (wid == 0) __builtin_amdgcn_s_wait_tensorcnt(0);
        __syncthreads();
    }

    v8f acc[2][2] = {};

    for (int k0 = 0, it = 0; k0 < K; k0 += 32, ++it) {
        const int cur = it & 1, nxt = cur ^ 1;
        const bool more = (k0 + 32) < K;

        // ---- kick off tile k0+32 (no waits: flies during the WMMA block) ----
        v8h breg;
        if (more) {
            const _Float16* src = bptr + (long)(k0 + 32) * bstep;
            breg = *(const v8h*)src;
            if (k0 + 64 < K) __builtin_prefetch(src + 32 * bstep, 0, 3);
            if (wid == 0) {
                unsigned long long ga = gbase + (unsigned long long)(k0 + 32) * 2ull;
                g0[1] = nxt ? lds1 : lds0;
                g0[2] = (unsigned)(ga & 0xffffffffull);
                g0[3] = (unsigned)((ga >> 32) & 0x1ffffffull) | (2u << 30);
                tdm_load(g0, g1);
            }
        }

        // ---- fragments from current buffer: pure b128 LDS loads + concat ----
        v16h af[2], bf[2];
#pragma unroll
        for (int mt = 0; mt < 2; ++mt) {
            const _Float16* pa = &As[cur][(wm * 32 + mt * 16 + (lane & 15)) * 40];
            v8h lo = *(const v8h*)(pa + ka);
            v8h hi = *(const v8h*)(pa + ka + 16);
            af[mt] = CONCAT16(lo, hi);
        }
#pragma unroll
        for (int nt = 0; nt < 2; ++nt) {
            const _Float16* pb = &Bs[cur][(wn * 32 + nt * 16 + (lane & 15)) * 40 + kb];
            v8h lo = *(const v8h*)(pb);
            v8h hi = *(const v8h*)(pb + 16);
            bf[nt] = CONCAT16(lo, hi);
        }
#pragma unroll
        for (int mt = 0; mt < 2; ++mt)
#pragma unroll
            for (int nt = 0; nt < 2; ++nt)
                acc[mt][nt] = __builtin_amdgcn_wmma_f32_16x16x32_f16(
                    false, af[mt], false, bf[nt], (short)0, acc[mt][nt], false, false);

        // ---- drain next tile into the alternate buffer and publish it ----
        if (more) {
            if (transB) {
                *(v8h*)&Bs[nxt][bn_ * 40 + bks] = breg;
            } else {
#pragma unroll
                for (int i = 0; i < 8; ++i) Bs[nxt][(bc_ + i) * 40 + bk_] = breg[i];
            }
            if (wid == 0) __builtin_amdgcn_s_wait_tensorcnt(0);
        }
        __syncthreads();
    }

    // ---- epilogue: C/D layout lanes0-15 M=r, lanes16-31 M=r+8, N=lane%16 ----
#pragma unroll
    for (int mt = 0; mt < 2; ++mt) {
        int rowb = bm0 + wm * 32 + mt * 16 + ((lane >> 4) << 3);
#pragma unroll
        for (int nt = 0; nt < 2; ++nt) {
            int col = bn0 + wn * 32 + nt * 16 + (lane & 15);
            float bv = hasBias ? bias[col] : 0.0f;
#pragma unroll
            for (int r = 0; r < 8; ++r) {
                long cidx = offC + (long)(rowb + r) * ldc + col;
                float v = acc[mt][nt][r] + bv;
                if (hasRes) v += resid[cidx];
                if (doRelu) v = fmaxf(v, 0.0f);
                if (outF16) ((_Float16*)Cv)[cidx] = (_Float16)v;
                else        ((float*)Cv)[cidx]    = v;
            }
        }
    }
}

// ============================================================
// x[t,:] = tok_emb[idx[t],:] + pos_emb[t % T,:]
// ============================================================
__global__ __launch_bounds__(256) void k_embed(
    const int* __restrict__ idx, const float* __restrict__ tok,
    const float* __restrict__ pos, float* __restrict__ x)
{
    int t = blockIdx.x, tid = threadIdx.x;
    const float* te = tok + (long)idx[t] * D_;
    const float* pe = pos + (long)(t % T_) * D_;
    float* xr = x + (long)t * D_;
#pragma unroll
    for (int j = 0; j < 4; ++j) {
        int i = tid + j * 256;
        xr[i] = te[i] + pe[i];
    }
}

// ============================================================
// LayerNorm over D, output f16 (feeds WMMA A operand)
// ============================================================
__global__ __launch_bounds__(256) void k_layernorm(
    const float* __restrict__ x, const float* __restrict__ g,
    const float* __restrict__ b, _Float16* __restrict__ y)
{
    __shared__ float r1[256], r2[256];
    int t = blockIdx.x, tid = threadIdx.x;
    const float* row = x + (long)t * D_;
    float s = 0.f, sq = 0.f, v[4];
#pragma unroll
    for (int j = 0; j < 4; ++j) {
        v[j] = row[tid + j * 256];
        s += v[j]; sq += v[j] * v[j];
    }
    r1[tid] = s; r2[tid] = sq; __syncthreads();
    for (int o = 128; o > 0; o >>= 1) {
        if (tid < o) { r1[tid] += r1[tid + o]; r2[tid] += r2[tid + o]; }
        __syncthreads();
    }
    float m   = r1[0] * (1.0f / D_);
    float var = r2[0] * (1.0f / D_) - m * m;
    float rstd = rsqrtf(var + 1e-5f);
    _Float16* yr = y + (long)t * D_;
#pragma unroll
    for (int j = 0; j < 4; ++j) {
        int i = tid + j * 256;
        yr[i] = (_Float16)((v[j] - m) * rstd * g[i] + b[i]);
    }
}

// ============================================================
// Causal softmax: reads raw scores at s (in d_out region), scales by 1/8,
// writes normalized probs back (f32) and f16 copy for PV GEMM.
// One block per (b,h,q) row of length T.
// ============================================================
__global__ __launch_bounds__(256) void k_softmax(
    float* __restrict__ s, _Float16* __restrict__ p16)
{
    __shared__ float red[256];
    long gid = blockIdx.x;            // z*T + q
    int q = (int)(gid % T_);
    float* row = s + gid * T_;
    _Float16* pr = p16 + gid * T_;
    int tid = threadIdx.x;
    const float scale = 0.125f;       // HS^-0.5

    float vals[4], lm = -3.0e38f;
#pragma unroll
    for (int j = 0; j < 4; ++j) {
        int k = tid + j * 256;
        vals[j] = (k <= q) ? row[k] * scale : -3.0e38f;
        lm = fmaxf(lm, vals[j]);
    }
    red[tid] = lm; __syncthreads();
    for (int o = 128; o > 0; o >>= 1) {
        if (tid < o) red[tid] = fmaxf(red[tid], red[tid + o]);
        __syncthreads();
    }
    float m = red[0]; __syncthreads();

    float e[4], ls = 0.f;
#pragma unroll
    for (int j = 0; j < 4; ++j) {
        int k = tid + j * 256;
        e[j] = (k <= q) ? expf(vals[j] - m) : 0.0f;
        ls += e[j];
    }
    red[tid] = ls; __syncthreads();
    for (int o = 128; o > 0; o >>= 1) {
        if (tid < o) red[tid] += red[tid + o];
        __syncthreads();
    }
    float rz = 1.0f / red[0];
#pragma unroll
    for (int j = 0; j < 4; ++j) {
        int k = tid + j * 256;
        float pv = e[j] * rz;
        row[k] = pv;
        pr[k] = (_Float16)pv;
    }
}

// ============================================================
// f32 -> f16 convert (grid-stride)
// ============================================================
__global__ void k_cvt(const float* __restrict__ s, _Float16* __restrict__ d, long n)
{
    long i = (long)blockIdx.x * blockDim.x + threadIdx.x;
    long st = (long)gridDim.x * blockDim.x;
    for (; i < n; i += st) d[i] = (_Float16)s[i];
}

// ============================================================
// Pack Wq/Wk/Wv [H,D,HS] f32 -> [D, H*HS] f16 (einsum 'btd,hds->bhts')
// ============================================================
__global__ __launch_bounds__(256) void k_pack_qkvw(
    const float* __restrict__ src, _Float16* __restrict__ dst)
{
    int i = blockIdx.x * 256 + threadIdx.x;   // grid 4096 -> D*D elements
    int d  = i >> 10;
    int n  = i & 1023;
    int h  = n >> 6;
    int sx = n & 63;
    dst[i] = (_Float16)src[((long)h * D_ + d) * HS_ + sx];
}

// ============================================================
// Streaming log-sum-exp over vocab chunks
// ============================================================
__global__ __launch_bounds__(256) void k_lse_init(
    float* __restrict__ m, float* __restrict__ s, float* __restrict__ t)
{
    int i = blockIdx.x * 256 + threadIdx.x;
    if (i < NT_) { m[i] = -3.0e38f; s[i] = 0.f; t[i] = 0.f; }
}

__global__ __launch_bounds__(256) void k_lse_update(
    const float* __restrict__ chunk, const int* __restrict__ targets,
    float* __restrict__ mbuf, float* __restrict__ sbuf, float* __restrict__ tbuf,
    int c0)
{
    __shared__ float red[256];
    int t = blockIdx.x, tid = threadIdx.x;
    const float* row = chunk + (long)t * CHUNK_;

    float lm = -3.0e38f;
    for (int j = tid; j < CHUNK_; j += 256) lm = fmaxf(lm, row[j]);
    red[tid] = lm; __syncthreads();
    for (int o = 128; o > 0; o >>= 1) {
        if (tid < o) red[tid] = fmaxf(red[tid], red[tid + o]);
        __syncthreads();
    }
    float cm = red[0]; __syncthreads();

    float oldm = mbuf[t];
    float newm = fmaxf(oldm, cm);
    float ls = 0.f;
    for (int j = tid; j < CHUNK_; j += 256) ls += expf(row[j] - newm);
    red[tid] = ls; __syncthreads();
    for (int o = 128; o > 0; o >>= 1) {
        if (tid < o) red[tid] += red[tid + o];
        __syncthreads();
    }
    if (tid == 0) {
        sbuf[t] = sbuf[t] * expf(oldm - newm) + red[0];
        mbuf[t] = newm;
        int tg = targets[t];
        if (tg >= c0 && tg < c0 + CHUNK_) tbuf[t] = row[tg - c0];
    }
}

__global__ __launch_bounds__(256) void k_loss(
    const float* __restrict__ mbuf, const float* __restrict__ sbuf,
    const float* __restrict__ tbuf, float* __restrict__ out)
{
    __shared__ float red[256];
    int tid = threadIdx.x;
    float s = 0.f;
    for (int t = tid; t < NT_; t += 256)
        s += mbuf[t] + logf(sbuf[t]) - tbuf[t];
    red[tid] = s; __syncthreads();
    for (int o = 128; o > 0; o >>= 1) {
        if (tid < o) red[tid] += red[tid + o];
        __syncthreads();
    }
    if (tid == 0) out[0] = red[0] * (1.0f / NT_);
}

// ============================================================
// host orchestration
// ============================================================
static inline void launch_gemm(hipStream_t s,
    const _Float16* A, const _Float16* B, void* C,
    const float* bias, const float* resid,
    int M, int N, int K, int lda, int ldb, int ldc, int nz,
    long sA1, long sA2, long sB1, long sB2, long sC1, long sC2, int flags)
{
    dim3 g(M / 128, N / 64, nz), b(256, 1, 1);
    k_gemm<<<g, b, 0, s>>>(A, B, C, bias, resid, lda, ldb, ldc, K,
                           sA1, sA2, sB1, sB2, sC1, sC2, flags);
}

extern "C" void kernel_launch(void* const* d_in, const int* in_sizes, int n_in,
                              void* d_out, int out_size, void* d_ws, size_t ws_size,
                              hipStream_t stream)
{
    (void)in_sizes; (void)n_in; (void)out_size; (void)ws_size;

    const int*   idx     = (const int*)d_in[0];
    const int*   targets = (const int*)d_in[1];
    const float* tok_emb = (const float*)d_in[2];
    const float* pos_emb = (const float*)d_in[3];
    const float* ln1_g   = (const float*)d_in[4];
    const float* ln1_b   = (const float*)d_in[5];
    const float* Wq      = (const float*)d_in[6];
    const float* Wk      = (const float*)d_in[7];
    const float* Wv      = (const float*)d_in[8];
    const float* Wp      = (const float*)d_in[9];
    const float* bp      = (const float*)d_in[10];
    const float* ln2_g   = (const float*)d_in[11];
    const float* ln2_b   = (const float*)d_in[12];
    const float* W1      = (const float*)d_in[13];
    const float* b1      = (const float*)d_in[14];
    const float* W2      = (const float*)d_in[15];
    const float* b2      = (const float*)d_in[16];
    const float* lnf_g   = (const float*)d_in[17];
    const float* lnf_b   = (const float*)d_in[18];
    const float* Wlm     = (const float*)d_in[19];
    const float* blm     = (const float*)d_in[20];

    float* loss_out = (float*)d_out;
    float* attn_out = loss_out + 1;   // [L,B,H,T,T] f32

    // ---- carve workspace (~380 MB) ----
    char* p = (char*)d_ws;
    auto bump = [&](size_t bytes) -> void* {
        void* r = (void*)p;
        p += (bytes + 255) & ~(size_t)255;
        return r;
    };
    float*    x0     = (float*)   bump((size_t)NT_ * D_ * 4);
    float*    x1     = (float*)   bump((size_t)NT_ * D_ * 4);
    _Float16* h16    = (_Float16*)bump((size_t)NT_ * D_ * 2);
    _Float16* w16    = (_Float16*)bump((size_t)D_ * FF_ * 2);      // biggest weight
    _Float16* q16    = (_Float16*)bump((size_t)NT_ * D_ * 2);
    _Float16* k16    = (_Float16*)bump((size_t)NT_ * D_ * 2);
    _Float16* v16    = (_Float16*)bump((size_t)NT_ * D_ * 2);
    _Float16* p16    = (_Float16*)bump((size_t)B_ * H_ * T_ * T_ * 2);
    _Float16* o16    = (_Float16*)bump((size_t)NT_ * D_ * 2);
    _Float16* ff16   = (_Float16*)bump((size_t)NT_ * FF_ * 2);
    _Float16* wlm16  = (_Float16*)bump((size_t)D_ * V_ * 2);
    float*    chunkb = (float*)   bump((size_t)NT_ * CHUNK_ * 4);
    float*    mbuf   = (float*)   bump((size_t)NT_ * 4);
    float*    sbuf   = (float*)   bump((size_t)NT_ * 4);
    float*    tbuf   = (float*)   bump((size_t)NT_ * 4);

    // ---- embedding ----
    k_embed<<<NT_, 256, 0, stream>>>(idx, tok_emb, pos_emb, x0);

    float* xc = x0;   // x always lives in xc at top of layer loop
    float* xn = x1;

    const long TT  = (long)T_ * T_;
    const long TD  = (long)T_ * D_;

    for (int l = 0; l < L_; ++l) {
        // ln1 -> h16
        k_layernorm<<<NT_, 256, 0, stream>>>(xc, ln1_g + l * D_, ln1_b + l * D_, h16);

        // Q, K, V projections  (f16 out, no bias)
        k_pack_qkvw<<<4096, 256, 0, stream>>>(Wq + (size_t)l * D_ * D_, w16);
        launch_gemm(stream, h16, w16, q16, nullptr, nullptr,
                    NT_, D_, D_, D_, D_, D_, 1, 0,0,0,0,0,0, 2);
        k_pack_qkvw<<<4096, 256, 0, stream>>>(Wk + (size_t)l * D_ * D_, w16);
        launch_gemm(stream, h16, w16, k16, nullptr, nullptr,
                    NT_, D_, D_, D_, D_, D_, 1, 0,0,0,0,0,0, 2);
        k_pack_qkvw<<<4096, 256, 0, stream>>>(Wv + (size_t)l * D_ * D_, w16);
        launch_gemm(stream, h16, w16, v16, nullptr, nullptr,
                    NT_, D_, D_, D_, D_, D_, 1, 0,0,0,0,0,0, 2);

        // scores = Q_bh @ K_bh^T  -> written raw into attn output region
        float* attn_l = attn_out + (size_t)l * B_ * H_ * T_ * T_;
        launch_gemm(stream, q16, k16, attn_l, nullptr, nullptr,
                    T_, T_, HS_, D_, D_, T_, B_ * H_,
                    TD, 64, TD, 64, (long)H_ * TT, TT, 1 /*transB*/);

        // causal softmax in place (f32) + f16 probs for PV
        k_softmax<<<B_ * H_ * T_, 256, 0, stream>>>(attn_l, p16);

        // O = P @ V  (f16 out into o16 laid out [b, q, h*HS+s])
        launch_gemm(stream, p16, v16, o16, nullptr, nullptr,
                    T_, HS_, T_, T_, D_, D_, B_ * H_,
                    (long)H_ * TT, TT, TD, 64, TD, 64, 2 /*outF16*/);

        // x = x + O @ Wp + bp   -> xn
        k_cvt<<<4096, 256, 0, stream>>>(Wp + (size_t)l * D_ * D_, w16, (long)D_ * D_);
        launch_gemm(stream, o16, w16, xn, bp + l * D_, xc,
                    NT_, D_, D_, D_, D_, D_, 1, 0,0,0,0,0,0, 8 | 16);

        // ln2 -> h16
        k_layernorm<<<NT_, 256, 0, stream>>>(xn, ln2_g + l * D_, ln2_b + l * D_, h16);

        // ff = relu(h @ W1 + b1)  (f16 out)
        k_cvt<<<8192, 256, 0, stream>>>(W1 + (size_t)l * D_ * FF_, w16, (long)D_ * FF_);
        launch_gemm(stream, h16, w16, ff16, b1 + l * FF_, nullptr,
                    NT_, FF_, D_, D_, FF_, FF_, 1, 0,0,0,0,0,0, 2 | 4 | 8);

        // x = x + ff @ W2 + b2   -> back into xc
        k_cvt<<<8192, 256, 0, stream>>>(W2 + (size_t)l * FF_ * D_, w16, (long)FF_ * D_);
        launch_gemm(stream, ff16, w16, xc, b2 + l * D_, xn,
                    NT_, D_, FF_, FF_, D_, D_, 1, 0,0,0,0,0,0, 8 | 16);
    }

    // ---- final LN + chunked logits with streaming logsumexp ----
    k_layernorm<<<NT_, 256, 0, stream>>>(xc, lnf_g, lnf_b, h16);
    k_cvt<<<16384, 256, 0, stream>>>(Wlm, wlm16, (long)D_ * V_);
    k_lse_init<<<NT_ / 256, 256, 0, stream>>>(mbuf, sbuf, tbuf);

    for (int c = 0; c < NCHUNK_; ++c) {
        int c0 = c * CHUNK_;
        launch_gemm(stream, h16, wlm16 + c0, chunkb, blm + c0, nullptr,
                    NT_, CHUNK_, D_, D_, V_, CHUNK_, 1, 0,0,0,0,0,0, 8);
        k_lse_update<<<NT_, 256, 0, stream>>>(chunkb, targets, mbuf, sbuf, tbuf, c0);
    }

    k_loss<<<1, 256, 0, stream>>>(mbuf, sbuf, tbuf, loss_out);
}
